// TripletLossWrapper_63848983822725
// MI455X (gfx1250) — matrix operand order repaired and checked by reading, not compiled, not run on hardware
//
#include <hip/hip_runtime.h>
#include <hip/hip_bf16.h>

#define NPT   384      // number of embeddings
#define DIM   256      // embedding dim
#define MARGINF 0.2f

typedef __attribute__((ext_vector_type(16))) _Float16 v16h;
typedef __attribute__((ext_vector_type(2)))  _Float16 h2;
typedef __attribute__((ext_vector_type(8)))  float    v8f;

// ---------------------------------------------------------------------------
// Kernel 1: row-normalize embeddings (f32 in) -> f16 normalized copy
// one block per row, 256 threads = one element each
// ---------------------------------------------------------------------------
__global__ void tl_normalize(const float* __restrict__ e, _Float16* __restrict__ eh) {
    __shared__ float red[DIM];
    const int row = blockIdx.x;
    const int t   = threadIdx.x;
    float v = e[row * DIM + t];
    red[t] = v * v;
    __syncthreads();
    #pragma unroll
    for (int s = DIM / 2; s > 0; s >>= 1) {
        if (t < s) red[t] += red[t + s];
        __syncthreads();
    }
    const float inv = 1.0f / fmaxf(sqrtf(red[0]), 1e-12f);
    eh[row * DIM + t] = (_Float16)(v * inv);
}

// ---------------------------------------------------------------------------
// Kernel 2: Gram via v_wmma_f32_16x16x32_f16, then D = sqrt(max(2 - 2g, 0))
// One wave per 16x16 tile of D. grid = (24, 24), block = 32 (EXEC all-ones).
//
// A fragment (16x32 f16, ISA layout): lane half selects K-octet;
//   VGPR v: K = (v>=4 ? 16:0) + half*8 + 2*(v%4), +1
// B fragment (32x16 f16): VGPR v: K = (v>=4 ? 16:0) + half*8 + 2*(v%4), +1; column N = lane%16.
// B column n == embedding row (tileN*16 + n), so both read the same buffer.
// ---------------------------------------------------------------------------
__global__ __launch_bounds__(32)
void tl_gram_dist(const _Float16* __restrict__ eh, float* __restrict__ D) {
    const int tileM = blockIdx.x;
    const int tileN = blockIdx.y;
    const int lane  = threadIdx.x;
    const int half  = lane >> 4;
    const int c     = lane & 15;

    const _Float16* rA = eh + (tileM * 16 + c) * DIM;
    const _Float16* rB = eh + (tileN * 16 + c) * DIM;

    v8f acc = {};
    for (int k0 = 0; k0 < DIM; k0 += 32) {
        v16h a, b;
        #pragma unroll
        for (int v = 0; v < 8; ++v) {
            const int ka = ((v & 4) ? 16 : 0) + half * 8 + ((v & 3) << 1);
            const h2 pa = *reinterpret_cast<const h2*>(rA + k0 + ka);
            a[2 * v]     = pa[0];
            a[2 * v + 1] = pa[1];
            const int kb = ((v & 4) ? 16 : 0) + half * 8 + ((v & 3) << 1);
            const h2 pb = *reinterpret_cast<const h2*>(rB + k0 + kb);
            b[2 * v]     = pb[0];
            b[2 * v + 1] = pb[1];
        }
        acc = __builtin_amdgcn_wmma_f32_16x16x32_f16(
            /*neg_a=*/false, a, /*neg_b=*/false, b,
            /*c_mod=*/(short)0, acc, /*reuse_a=*/false, /*reuse_b=*/false);
    }

    // C/D layout: VGPR r -> M = half*8 + r, N = c
    #pragma unroll
    for (int r = 0; r < 8; ++r) {
        const int m  = half * 8 + r;
        const float d2 = fmaxf(2.0f - 2.0f * acc[r], 0.0f);
        const float d  = (d2 > 0.0f) ? sqrtf(d2) : 0.0f;
        D[(tileM * 16 + m) * NPT + tileN * 16 + c] = d;
    }
}

// ---------------------------------------------------------------------------
// Kernel 3: semi-hard triplet scan. One block per anchor i; the anchor's
// distance row + labels staged in LDS; threads stride over negatives.
// semihard: 0 < tm <= MARGIN, loss = MARGIN - tm; nonzero iff tm < MARGIN.
// ---------------------------------------------------------------------------
__global__ void tl_triplets(const float* __restrict__ D, const int* __restrict__ labels,
                            float* __restrict__ psum, unsigned* __restrict__ pcnt) {
    __shared__ float    Drow[NPT];
    __shared__ int      lab[NPT];
    __shared__ float    rsum[256];
    __shared__ unsigned rcnt[256];

    const int i = blockIdx.x;
    const int t = threadIdx.x;

    for (int idx = t; idx < NPT; idx += 256) {
        Drow[idx] = D[i * NPT + idx];
        lab[idx]  = labels[idx];
    }
    __syncthreads();

    const int li = lab[i];
    float    sum = 0.0f;
    unsigned cnt = 0u;
    for (int j = 0; j < NPT; ++j) {
        if (j == i || lab[j] != li) continue;    // positive mask
        const float ap = Drow[j];
        for (int k = t; k < NPT; k += 256) {
            if (lab[k] == li) continue;          // negative mask
            const float tm = Drow[k] - ap;       // an - ap
            if (tm > 0.0f && tm < MARGINF) {
                sum += MARGINF - tm;
                cnt += 1u;
            }
        }
    }

    rsum[t] = sum;
    rcnt[t] = cnt;
    __syncthreads();
    #pragma unroll
    for (int s = 128; s > 0; s >>= 1) {
        if (t < s) { rsum[t] += rsum[t + s]; rcnt[t] += rcnt[t + s]; }
        __syncthreads();
    }
    if (t == 0) { psum[i] = rsum[0]; pcnt[i] = rcnt[0]; }
}

// ---------------------------------------------------------------------------
// Kernel 4: final reduction -> scalar mean over nonzero losses
// ---------------------------------------------------------------------------
__global__ void tl_finalize(const float* __restrict__ psum, const unsigned* __restrict__ pcnt,
                            float* __restrict__ out) {
    __shared__ float    rs[256];
    __shared__ unsigned rc[256];
    const int t = threadIdx.x;
    float    s = 0.0f;
    unsigned c = 0u;
    for (int idx = t; idx < NPT; idx += 256) { s += psum[idx]; c += pcnt[idx]; }
    rs[t] = s;
    rc[t] = c;
    __syncthreads();
    #pragma unroll
    for (int st = 128; st > 0; st >>= 1) {
        if (t < st) { rs[t] += rs[t + st]; rc[t] += rc[t + st]; }
        __syncthreads();
    }
    if (t == 0) out[0] = (rc[0] > 0u) ? rs[0] / (float)rc[0] : 0.0f;
}

// ---------------------------------------------------------------------------
// Launch: normalize -> WMMA Gram/dist -> triplet partials -> finalize
// Workspace layout (bytes):
//   [0,         196608)  f16 normalized embeddings (384*256*2)
//   [196608,    786432)  f32 distance matrix D     (384*384*4)
//   [786432,    787968)  f32 per-anchor loss sums  (384*4)
//   [787968,    789504)  u32 per-anchor counts     (384*4)
// ---------------------------------------------------------------------------
extern "C" void kernel_launch(void* const* d_in, const int* in_sizes, int n_in,
                              void* d_out, int out_size, void* d_ws, size_t ws_size,
                              hipStream_t stream) {
    const float* emb    = (const float*)d_in[0];
    const int*   labels = (const int*)d_in[1];
    char* ws = (char*)d_ws;

    _Float16* eh   = (_Float16*)(ws);
    float*    D    = (float*)(ws + 196608);
    float*    psum = (float*)(ws + 786432);
    unsigned* pcnt = (unsigned*)(ws + 787968);

    tl_normalize<<<NPT, DIM, 0, stream>>>(emb, eh);

    dim3 gtiles(NPT / 16, NPT / 16);   // 24 x 24 tiles, one wave each
    tl_gram_dist<<<gtiles, 32, 0, stream>>>(eh, D);

    tl_triplets<<<NPT, 256, 0, stream>>>(D, labels, psum, pcnt);

    tl_finalize<<<1, 256, 0, stream>>>(psum, pcnt, (float*)d_out);
}
